// FPModule_33397665694063
// MI455X (gfx1250) — matrix operand order repaired and checked by reading, not compiled, not run on hardware
//
#include <hip/hip_runtime.h>
#include <hip/hip_bf16.h>

// ---------------------------------------------------------------------------
// FPModule (PointNet++ feature propagation) for MI455X / gfx1250, wave32.
// knn(K=3) -> inverse-dist interp -> concat -> [512x512] GEMM+ReLU ->
// [512x256] GEMM+ReLU.  GEMMs: v_wmma_f32_16x16x32_f16, B tile staged in LDS
// via async global->LDS (ASYNCcnt), A stream register double-buffered.
// ---------------------------------------------------------------------------

typedef __attribute__((ext_vector_type(16))) _Float16 v16h;
typedef __attribute__((ext_vector_type(8)))  float    v8f;

#define NFINE   65536      // B*M
#define NCOARSE 4096       // N per cloud
#define CIN     256
#define CH0     512
#define CH1     512
#define CH2     256

#define USE_ASYNC_LDS 1    // CDNA5 async global->LDS staging path

// A-matrix 16x32 f16 lane/K mapping (ISA 7.12.2):
// lane<16 : M=lane,    halves i<8 -> K=i,     i>=8 -> K=i+8   (0..7,16..23)
// lane>=16: M=lane-16, halves i<8 -> K=i+8,   i>=8 -> K=i+16  (8..15,24..31)
__device__ __forceinline__ int mapK(int lane, int i) {
  int base = (lane < 16) ? 0 : 8;
  return base + ((i < 8) ? i : (i + 8));
}

// ---------------------------------------------------------------------------
// Kernel 1: brute-force 3-NN per fine point, coarse cloud tiled through LDS.
// ---------------------------------------------------------------------------
__global__ __launch_bounds__(256)
void knn_kernel(const float* __restrict__ pos, const float* __restrict__ pos_skip,
                int* __restrict__ idxOut, float* __restrict__ wOut) {
  __shared__ float sp[256][3];
  const int q = blockIdx.x * 256 + threadIdx.x;   // fine point id
  const int b = q >> 14;                          // 16384 fine pts per cloud
  const float qx = pos_skip[q * 3 + 0];
  const float qy = pos_skip[q * 3 + 1];
  const float qz = pos_skip[q * 3 + 2];
  float d0 = 1e30f, d1 = 1e30f, d2 = 1e30f;
  int   i0 = 0,     i1 = 0,     i2 = 0;
  for (int t = 0; t < NCOARSE / 256; ++t) {
    const int pi = (b << 12) + t * 256 + threadIdx.x;
    sp[threadIdx.x][0] = pos[pi * 3 + 0];
    sp[threadIdx.x][1] = pos[pi * 3 + 1];
    sp[threadIdx.x][2] = pos[pi * 3 + 2];
    __syncthreads();
    for (int j = 0; j < 256; ++j) {               // uniform j -> LDS broadcast
      const float dx = qx - sp[j][0];
      const float dy = qy - sp[j][1];
      const float dz = qz - sp[j][2];
      const float d = dx * dx + dy * dy + dz * dz;
      const int gi = (b << 12) + t * 256 + j;
      if (d < d2) {
        if (d < d0)      { d2 = d1; i2 = i1; d1 = d0; i1 = i0; d0 = d; i0 = gi; }
        else if (d < d1) { d2 = d1; i2 = i1; d1 = d;  i1 = gi; }
        else             { d2 = d;  i2 = gi; }
      }
    }
    __syncthreads();
  }
  const float w0 = 1.0f / fmaxf(d0, 1e-16f);
  const float w1 = 1.0f / fmaxf(d1, 1e-16f);
  const float w2 = 1.0f / fmaxf(d2, 1e-16f);
  const float wsum = w0 + w1 + w2;
  wOut[q * 3 + 0] = w0 / wsum;
  wOut[q * 3 + 1] = w1 / wsum;
  wOut[q * 3 + 2] = w2 / wsum;
  idxOut[q * 3 + 0] = i0;
  idxOut[q * 3 + 1] = i1;
  idxOut[q * 3 + 2] = i2;
}

// ---------------------------------------------------------------------------
// Kernel 2: interpolate + concat skip, writing H [65536 x 512] f16 directly in
// WMMA A-fragment-packed layout: chunk = ((rowTile*16 + kTile)*32 + lane).
// ---------------------------------------------------------------------------
__global__ __launch_bounds__(256)
void interp_pack_kernel(const float* __restrict__ x, const float* __restrict__ xs,
                        const int* __restrict__ idx, const float* __restrict__ w,
                        _Float16* __restrict__ Hp) {
  const int t = blockIdx.x * 256 + threadIdx.x;   // one 32B chunk per thread
  const int lane    = t & 31;
  const int kTile   = (t >> 5) & 15;              // 512/32 k-tiles
  const int rowTile = t >> 9;
  const int row = rowTile * 16 + (lane & 15);
  const int j0 = idx[row * 3 + 0];
  const int j1 = idx[row * 3 + 1];
  const int j2 = idx[row * 3 + 2];
  const float w0 = w[row * 3 + 0];
  const float w1 = w[row * 3 + 1];
  const float w2 = w[row * 3 + 2];
  v16h out;
  if (kTile < 8) {                                // k in [0,256): interpolated
    #pragma unroll
    for (int i = 0; i < 16; ++i) {
      const int k = kTile * 32 + mapK(lane, i);
      const float v = w0 * x[(size_t)j0 * CIN + k] +
                      w1 * x[(size_t)j1 * CIN + k] +
                      w2 * x[(size_t)j2 * CIN + k];
      out[i] = (_Float16)v;
    }
  } else {                                        // k in [256,512): skip feats
    #pragma unroll
    for (int i = 0; i < 16; ++i) {
      const int k = kTile * 32 + mapK(lane, i);
      out[i] = (_Float16)xs[(size_t)row * 256 + (k - 256)];
    }
  }
  ((v16h*)Hp)[t] = out;
}

// ---------------------------------------------------------------------------
// Kernel 3: pack row-major f32 weights [K x N] into WMMA B-fragment layout.
// B 32x16 f16: lane%16 = column, lane/16 selects K set (0..7,16..23 / 8..15,24..31).
// ---------------------------------------------------------------------------
__global__ __launch_bounds__(256)
void pack_w_kernel(const float* __restrict__ W, _Float16* __restrict__ Bp, int Ndim) {
  const int t = blockIdx.x * 256 + threadIdx.x;
  const int lane = t & 31;
  const int nt   = (t >> 5) % (Ndim >> 4);
  const int kt   = (t >> 5) / (Ndim >> 4);
  const int col   = (nt << 4) + (lane & 15);
  const int kbase = (kt << 5) + ((lane >> 4) << 3);
  v16h out;
  #pragma unroll
  for (int i = 0; i < 16; ++i)
    out[i] = (_Float16)W[(size_t)(kbase + ((i < 8) ? i : (i + 8))) * Ndim + col];
  ((v16h*)Bp)[t] = out;
}

// ---------------------------------------------------------------------------
// Kernel 4/5: GEMM + bias + ReLU.
//  Block: 8 waves, tile 256 rows x 64 cols. Wave: 32 rows x 64 cols
//  (2 A frags, 4 B frags, 8 WMMAs per k-step; K=512 in 16 steps).
//  B tile (512x64 f16 = 64KB) staged in LDS once per block via async
//  global->LDS; A fragments register double-buffered (prefetch next k-step).
//  PACK_OUT: repack ReLU output to A-fragment f16 (reusing the B-stage LDS).
// ---------------------------------------------------------------------------
template <int NDIM, bool PACK_OUT>
__global__ __launch_bounds__(256)
void gemm_kernel(const _Float16* __restrict__ Ap, const _Float16* __restrict__ Bp,
                 const float* __restrict__ bias, _Float16* __restrict__ Cp,
                 float* __restrict__ Cf) {
  constexpr int KT   = CH0 / 32;     // 16 k-steps (K=512 for both layers)
  constexpr int NT16 = NDIM / 16;    // B n-tiles in packed weight tensor
  const int lane = threadIdx.x & 31;
  const int wave = threadIdx.x >> 5;
  const int tid  = threadIdx.x;
  const int rowTile0 = blockIdx.x * 16 + wave * 2;   // two 16-row tiles / wave
  const int c0 = blockIdx.y * 64;
  const int nb = c0 >> 4;

  // ---- Stage B tile: 64 fragments x (32 lanes x 32B) = 64KB into LDS -------
  __shared__ __align__(32) _Float16 sB[KT * 4 * 32 * 16];   // 64KB
  for (int f = tid; f < KT * 4 * 32; f += 256) {            // 2048 chunks
    const int frag = f >> 5;                                 // kt*4 + j
    const int l2   = f & 31;
    const int kt   = frag >> 2;
    const int j    = frag & 3;
    const _Float16* src =
        Bp + (((size_t)kt * NT16 + nb + j) * 32 + l2) * 16;
#if USE_ASYNC_LDS
    typedef __attribute__((address_space(3))) _Float16 lds_h;
    const unsigned ldsAddr =
        (unsigned)(unsigned long long)(lds_h*)&sB[(size_t)f * 16];
    const unsigned long long ga = (unsigned long long)src;
    asm volatile(
        "global_load_async_to_lds_b128 %0, %1, off\n\t"
        "global_load_async_to_lds_b128 %0, %1, off offset:16"
        :: "v"(ldsAddr), "v"(ga) : "memory");
#else
    ((v16h*)sB)[f] = *(const v16h*)src;
#endif
  }
#if USE_ASYNC_LDS
  asm volatile("s_wait_asynccnt 0x0" ::: "memory");
#endif
  __syncthreads();

  // ---- Mainloop: A register double-buffer + LDS B broadcast ----------------
  const v16h* Av  = (const v16h*)Ap;
  const v16h* sBv = (const v16h*)sB;
  const v16h* a0p = Av + ((size_t)(rowTile0 + 0) * KT) * 32 + lane;
  const v16h* a1p = Av + ((size_t)(rowTile0 + 1) * KT) * 32 + lane;

  v8f acc[2][4] = {};
  v16h a0 = a0p[0];
  v16h a1 = a1p[0];

  #pragma unroll 2
  for (int kt = 0; kt < KT; ++kt) {
    const int ktn = (kt + 1) & (KT - 1);       // wraps on last iter (harmless)
    const v16h a0n = a0p[(size_t)ktn * 32];
    const v16h a1n = a1p[(size_t)ktn * 32];
    #pragma unroll
    for (int j = 0; j < 4; ++j) {
      const v16h b = sBv[(size_t)(kt * 4 + j) * 32 + lane];
      acc[0][j] = __builtin_amdgcn_wmma_f32_16x16x32_f16(
          false, a0, false, b, (short)0, acc[0][j], false, false);
      acc[1][j] = __builtin_amdgcn_wmma_f32_16x16x32_f16(
          false, a1, false, b, (short)0, acc[1][j], false, false);
    }
    a0 = a0n;
    a1 = a1n;
  }

  const int rlo = lane & 15;
  const int hi  = lane >> 4;           // C/D: lanes 16-31 hold rows M+8

  if constexpr (PACK_OUT) {
    // Reuse the 64KB B-stage LDS as the repack staging buffer (36KB needed).
    __syncthreads();                   // everyone done reading sB
    _Float16* sc = sB + (size_t)wave * 32 * 72;   // 32 rows x 72-half stride
    #pragma unroll
    for (int rt = 0; rt < 2; ++rt) {
      #pragma unroll
      for (int j = 0; j < 4; ++j) {
        #pragma unroll
        for (int r = 0; r < 8; ++r) {
          const int row = rt * 16 + (hi ? (r + 8) : r);
          const int col = j * 16 + rlo;
          float v = acc[rt][j][r] + bias[c0 + col];
          v = v > 0.0f ? v : 0.0f;
          sc[row * 72 + col] = (_Float16)v;
        }
      }
    }
    __syncthreads();
    constexpr int OKT = NDIM / 32;     // next layer's k-tiles
    #pragma unroll
    for (int rt = 0; rt < 2; ++rt) {
      #pragma unroll
      for (int kt2 = 0; kt2 < 2; ++kt2) {
        v16h outv;
        const int base = kt2 * 32 + (hi ? 8 : 0);
        #pragma unroll
        for (int i = 0; i < 16; ++i) {
          const int kk = base + ((i < 8) ? i : (i + 8));
          outv[i] = sc[(rt * 16 + rlo) * 72 + kk];
        }
        const size_t cidx =
            ((size_t)(rowTile0 + rt) * OKT + (c0 >> 5) + kt2) * 32 + lane;
        ((v16h*)Cp)[cidx] = outv;
      }
    }
  } else {
    #pragma unroll
    for (int rt = 0; rt < 2; ++rt) {
      #pragma unroll
      for (int j = 0; j < 4; ++j) {
        #pragma unroll
        for (int r = 0; r < 8; ++r) {
          const int row = (rowTile0 + rt) * 16 + (hi ? (r + 8) : r);
          const int col = c0 + j * 16 + rlo;
          float v = acc[rt][j][r] + bias[col];
          v = v > 0.0f ? v : 0.0f;
          Cf[(size_t)row * NDIM + col] = v;
        }
      }
    }
  }
}

// ---------------------------------------------------------------------------
// Kernel 6: copy pos_skip + batch_skip into the output tail.
// ---------------------------------------------------------------------------
__global__ __launch_bounds__(256)
void tail_kernel(const float* __restrict__ pos_skip, float* __restrict__ out,
                 int batchMode) {
  const int t = blockIdx.x * 256 + threadIdx.x;
  const size_t HOFF = (size_t)NFINE * CH2;           // 16,777,216
  if (t < NFINE * 3) out[HOFF + t] = pos_skip[t];
  if (t < NFINE) {
    const int b = t >> 14;
    if (batchMode == 1)      out[HOFF + NFINE * 3 + t] = (float)b;
    else if (batchMode == 2) ((long long*)(out + HOFF + NFINE * 3))[t] = (long long)b;
  }
}

// ---------------------------------------------------------------------------
extern "C" void kernel_launch(void* const* d_in, const int* in_sizes, int n_in,
                              void* d_out, int out_size, void* d_ws, size_t ws_size,
                              hipStream_t stream) {
  const float* x        = (const float*)d_in[0];
  const float* pos      = (const float*)d_in[1];
  const float* x_skip   = (const float*)d_in[3];
  const float* pos_skip = (const float*)d_in[4];
  const float* W1       = (const float*)d_in[6];
  const float* b1       = (const float*)d_in[7];
  const float* W2       = (const float*)d_in[8];
  const float* b2       = (const float*)d_in[9];
  float* outF = (float*)d_out;

  // Workspace layout (~137 MB total)
  char* ws = (char*)d_ws;
  int*       idx = (int*)(ws);                              //  0.75 MB
  float*     w   = (float*)(ws + 786432);                   //  0.75 MB
  _Float16*  Hp  = (_Float16*)(ws + 1572864);               // 64 MB packed H
  _Float16*  W1p = (_Float16*)(ws + 1572864 + 67108864);    // 0.5 MB
  _Float16*  W2p = W1p + (size_t)CH0 * CH1;                 // 0.25 MB
  _Float16*  C1p = W2p + (size_t)CH1 * CH2;                 // 64 MB packed C1

  knn_kernel<<<dim3(NFINE / 256), dim3(256), 0, stream>>>(pos, pos_skip, idx, w);

  interp_pack_kernel<<<dim3(8192), dim3(256), 0, stream>>>(x, x_skip, idx, w, Hp);

  pack_w_kernel<<<dim3((16 * 32 * 32) / 256), dim3(256), 0, stream>>>(W1, W1p, CH1);
  pack_w_kernel<<<dim3((16 * 16 * 32) / 256), dim3(256), 0, stream>>>(W2, W2p, CH2);

  // GEMM1: 65536x512 @ 512x512, ReLU, repacked f16 output
  gemm_kernel<CH1, true><<<dim3(256, CH1 / 64), dim3(256), 0, stream>>>(
      Hp, W1p, b1, C1p, nullptr);
  // GEMM2: 65536x512 @ 512x256, ReLU, f32 output -> d_out
  gemm_kernel<CH2, false><<<dim3(256, CH2 / 64), dim3(256), 0, stream>>>(
      C1p, W2p, b2, nullptr, outF);

  // Tail: pos_skip + batch_skip (mode inferred from out_size)
  const long long HOFF = (long long)NFINE * CH2;
  int batchMode = 0;
  if ((long long)out_size >= HOFF + NFINE * 3 + NFINE * 2) batchMode = 2;  // int64
  else if ((long long)out_size >= HOFF + NFINE * 3 + NFINE) batchMode = 1; // as f32
  tail_kernel<<<dim3((NFINE * 3) / 256), dim3(256), 0, stream>>>(pos_skip, outF, batchMode);
}
